// CompressSensory_56805237457582
// MI455X (gfx1250) — matrix operand order repaired and checked by reading, not compiled, not run on hardware
//
#include <hip/hip_runtime.h>
#include <hip/hip_bf16.h>
#include <stdint.h>

typedef __attribute__((ext_vector_type(16))) _Float16 v16h;
typedef __attribute__((ext_vector_type(8)))  float    v8f;

#define X_DIM  45
#define XC_DIM 10
#define ROWS_PER_BLOCK 256   // 8 waves * 2 tiles * 16 rows
#define THREADS 256

// ---- CDNA5 async global->LDS copy (gfx1250, tracked by ASYNCcnt) ----
__device__ __forceinline__ void async_ld_b128(uint32_t lds_off, uint64_t gaddr) {
  asm volatile("global_load_async_to_lds_b128 %0, %1, off"
               :: "v"(lds_off), "v"(gaddr) : "memory");
}
__device__ __forceinline__ void async_ld_b32(uint32_t lds_off, uint64_t gaddr) {
  asm volatile("global_load_async_to_lds_b32 %0, %1, off"
               :: "v"(lds_off), "v"(gaddr) : "memory");
}
__device__ __forceinline__ void wait_async0() {
  asm volatile("s_wait_asynccnt 0" ::: "memory");
}

__global__ __launch_bounds__(THREADS)
void CompressSensory_argmax_twohot_wmma(const float* __restrict__ x,
                                        const float* __restrict__ table,
                                        float* __restrict__ out) {
  __shared__ float sx[ROWS_PER_BLOCK * X_DIM];          // 46080 B staged x rows
  __shared__ __align__(32) _Float16 sBT[16 * 64];       // 2048 B: sBT[n*64+k] = tablePad[k][n]

  const int t = threadIdx.x;
  const float* gsrc = x + (size_t)blockIdx.x * (ROWS_PER_BLOCK * X_DIM);

  // ---- Stage 1a: async coalesced copy of 256 rows (11520 floats) into LDS.
  // 11 full b128 rounds cover floats [0, 11264); exact fit, no guards.
  #pragma unroll
  for (int i = 0; i < 11; ++i) {
    int f = (t + i * THREADS) * 4;   // float index, 16B aligned
    async_ld_b128((uint32_t)(uintptr_t)&sx[f], (uint64_t)(uintptr_t)(gsrc + f));
  }
  // exact b32 tail: floats [11264, 11520) = 256 floats = 1 per thread, no guard.
  {
    int f = 11264 + t;
    async_ld_b32((uint32_t)(uintptr_t)&sx[f], (uint64_t)(uintptr_t)(gsrc + f));
  }

  // ---- Stage 1b: zero-padded, transposed f16 table in LDS (once per block).
  #pragma unroll
  for (int i = t; i < 16 * 64; i += THREADS) {          // 4 iterations
    int n = i >> 6;                                     // 0..15 (output column)
    int k = i & 63;                                     // 0..63 (padded K)
    int kc = (k < X_DIM)  ? k : (X_DIM - 1);
    int nc = (n < XC_DIM) ? n : (XC_DIM - 1);
    float v = table[kc * XC_DIM + nc];                  // always in-bounds
    sBT[i] = (k < X_DIM && n < XC_DIM) ? (_Float16)v : (_Float16)0.0f;
  }

  wait_async0();          // my wave's async LDS writes complete
  __syncthreads();        // all staging (sx + sBT) visible

  const int lane = t & 31;
  const int w    = t >> 5;     // wave id 0..7 -> rows [w*32, w*32+32) as two 16-row tiles
  const int m    = lane & 15;  // A-matrix row M / D-matrix column N
  const int g    = lane >> 4;  // half-wave select

  // B fragments loaded once, reused by both tiles (element e -> K = e+16g, N = m).
  v16h b1 = *(const v16h*)(&sBT[m * 64 + 16 * g]);
  v16h b2 = *(const v16h*)(&sBT[m * 64 + 32 + 16 * g]);

  #pragma unroll
  for (int tile = 0; tile < 2; ++tile) {
    // ---- Stage 2: per-row argmax, two lanes per row ----
    const float* row = &sx[(w * 32 + tile * 16 + m) * X_DIM];
    const int kbase = 22 * g;          // g=0: [0,22]; g=1: [22,44] (overlap harmless)
    float best = row[kbase];
    int   bidx = kbase;
    #pragma unroll
    for (int j = 1; j < 23; ++j) {     // uniform, unrolled -> static ds_load clause
      float v = row[kbase + j];
      if (v > best) { best = v; bidx = kbase + j; }  // strict '>' keeps first occurrence
    }
    float obest = __shfl_xor(best, 16, 32);
    int   oidx  = __shfl_xor(bidx, 16, 32);
    if (obest > best || (obest == best && oidx < bidx)) { best = obest; bidx = oidx; }
    // both lanes m, m+16 now hold argmax index of row m

    // ---- Stage 3: build one-hot A fragments analytically (dword-granular).
    // Lane's a1 covers K in [16g,16g+16) at e=d, d = bidx-16g.
    int d1 = bidx - 16 * g;
    int e1 = ((unsigned)d1 < 16u) ? d1 : -1;
    int d2 = d1 - 32;                  // same map for the K+32 fragment
    int e2 = ((unsigned)d2 < 16u) ? d2 : -1;
    uint32_t v1 = 0x3C00u << ((e1 & 1) << 4);   // f16 1.0 in the right half
    uint32_t v2 = 0x3C00u << ((e2 & 1) << 4);
    int w1 = e1 >> 1;                  // e1==-1 -> w1==-1 -> fragment stays zero
    int w2 = e2 >> 1;

    union F8 { uint32_t u[8]; v16h h; } A1, A2;
    #pragma unroll
    for (int i = 0; i < 8; ++i) {
      A1.u[i] = (w1 == i) ? v1 : 0u;
      A2.u[i] = (w2 == i) ? v2 : 0u;
    }

    // ---- WMMA: D(16x16) = onehot(16x64) @ tablePad(64x16) ----
    v8f c = {};
    c = __builtin_amdgcn_wmma_f32_16x16x32_f16(false, A1.h, false, b1, (short)0, c, false, false);
    c = __builtin_amdgcn_wmma_f32_16x16x32_f16(false, A2.h, false, b2, (short)0, c, false, false);

    // ---- Stage 4: store D. Lane holds D[M = r + 8g, N = m] in c[r]. ----
    const size_t rowBase = (size_t)blockIdx.x * ROWS_PER_BLOCK + (size_t)w * 32 + tile * 16;
    if (m < XC_DIM) {
      #pragma unroll
      for (int r = 0; r < 8; ++r) {
        int M = r + 8 * g;
        out[(rowBase + M) * XC_DIM + m] = c[r];
      }
    }
  }
}

extern "C" void kernel_launch(void* const* d_in, const int* in_sizes, int n_in,
                              void* d_out, int out_size, void* d_ws, size_t ws_size,
                              hipStream_t stream) {
  const float* x     = (const float*)d_in[0];   // (BATCH, 45) f32
  const float* table = (const float*)d_in[1];   // (45, 10)   f32
  float* out = (float*)d_out;                   // (BATCH, 10) f32
  (void)d_ws; (void)ws_size; (void)n_in; (void)out_size;

  int n_rows = in_sizes[0] / X_DIM;             // 1048576
  int blocks = n_rows / ROWS_PER_BLOCK;         // 4096 (exact fit)
  CompressSensory_argmax_twohot_wmma<<<blocks, THREADS, 0, stream>>>(x, table, out);
}
